// Baseline_net_83193516523924
// MI455X (gfx1250) — matrix operand order repaired and checked by reading, not compiled, not run on hardware
//
#include <hip/hip_runtime.h>
#include <hip/hip_bf16.h>

typedef __attribute__((ext_vector_type(16))) _Float16 v16h;
typedef __attribute__((ext_vector_type(8)))  _Float16 v8h;
typedef __attribute__((ext_vector_type(8)))  float    v8f;

__device__ __forceinline__ int reflect1(int i, int n) {
    if (i < 0) return -i;
    if (i >= n) return 2 * n - 2 - i;
    return i;
}
__device__ __forceinline__ int clampi(int i, int n) {
    return i < 0 ? 0 : (i >= n ? n - 1 : i);
}

// ---------------------------------------------------------------------------
// Pre-pack conv weights: f32 [Cout][Cin][3][3] -> f16 [tap][Cout][Cin]
// (K innermost, so conv staging is a pure 16B-aligned byte copy -> async DMA).
// ---------------------------------------------------------------------------
__global__ __launch_bounds__(256) void pack_w_kernel(
    const float* __restrict__ w, _Float16* __restrict__ wp, int Cout, int Cin)
{
    int idx = blockIdx.x * 256 + threadIdx.x;
    int total = Cout * Cin * 9;
    if (idx >= total) return;
    int c   = idx % Cin;
    int t2  = idx / Cin;
    int m   = t2 % Cout;
    int tap = t2 / Cout;
    wp[idx] = (_Float16)w[((size_t)m * Cin + c) * 9 + tap];
}

// ---------------------------------------------------------------------------
// Implicit-GEMM 3x3 conv (reflect pad 1) via WMMA f16 -> f32.
// Block: 256 threads = 8 waves. Tile: 64 out-channels x 64 pixels
// (2 rows x 32 cols). Each wave holds one A fragment and two accumulators
// (row y0 / y0+1), so weights are amortized over 2x the output.
// grid.x = (W/32)*(H/2), grid.y = Cout/64, grid.z = B
// LDS puts K (cin) innermost with 40-half (80B, 16B-aligned) stride so every
// WMMA fragment is exactly 2x ds_load_b128. Weight slab is staged with
// GLOBAL_LOAD_ASYNC_TO_LDS_B128 (ASYNCcnt), overlapping the f32->f16 input
// conversion path.
// ---------------------------------------------------------------------------
#define KP 40   // padded cin stride in halves (80 B, multiple of 16 B)

__global__ __launch_bounds__(256) void conv3x3_wmma_kernel(
    const float* __restrict__ in, const _Float16* __restrict__ wpk,
    const float* __restrict__ bias, float* __restrict__ out,
    int Cin, int Cout, int H, int W, int do_relu)
{
    const int tid  = threadIdx.x;
    const int lane = tid & 31;
    const int wave = tid >> 5;
    const int msub = wave & 3;   // 0..3 -> which 16 of the 64 out channels
    const int nsub = wave >> 2;  // 0..1 -> which 16 of the 32 columns

    const int segs_per_row = W >> 5;
    const int y0 = (blockIdx.x / segs_per_row) * 2;
    const int x0 = (blockIdx.x % segs_per_row) << 5;
    const int m0 = blockIdx.y << 6;
    const int b  = blockIdx.z;

    __shared__ _Float16 sIn[4][34][KP];   // [row(y0-1..y0+2)][col(34)][cin(32)]
    __shared__ _Float16 sW[9][64][KP];    // [tap][m][cin(32)]

    v8f acc0 = {};   // row y0
    v8f acc1 = {};   // row y0+1

    const int hi = lane >> 4;   // lane half
    const int lm = lane & 15;

    for (int cin0 = 0; cin0 < Cin; cin0 += 32) {
        // ---- stage weight slab asynchronously: 9 taps x 64 m x 32 cin ----
        // Pre-packed f16, K innermost in both layouts: pure 16B block copies.
        // 2304 chunks / 256 threads = 9 async b128 per thread, no divergence.
        for (int i = tid; i < (9 * 64 * 32) / 8; i += 256) {
            int tm  = i >> 2;          // tap*64 + m
            int c8  = (i & 3) * 8;     // which 8 of the 32 cin
            int tap = tm >> 6;
            int m   = tm & 63;
            unsigned lds_off = (unsigned)(uintptr_t)&sW[tap][m][c8];
            unsigned long long ga = (unsigned long long)(uintptr_t)
                &wpk[((size_t)tap * Cout + m0 + m) * Cin + cin0 + c8];
            asm volatile("global_load_async_to_lds_b128 %0, %1, off"
                         :: "v"(lds_off), "v"(ga) : "memory");
        }
        // ---- stage input patch: 32 cin x 4 rows x 34 cols, f32 -> f16 ----
        // (col fastest within a thread-step => coalesced HBM reads)
        for (int i = tid; i < 32 * 4 * 34; i += 256) {
            int c   = i / (4 * 34);
            int rem = i % (4 * 34);
            int r   = rem / 34;
            int col = rem % 34;
            int ry  = reflect1(y0 - 1 + r, H);
            int gx  = reflect1(x0 - 1 + col, W);
            sIn[r][col][c] =
                (_Float16)in[(((size_t)b * Cin + cin0 + c) * H + ry) * W + gx];
        }
        asm volatile("s_wait_asynccnt 0x0" ::: "memory");
        __syncthreads();

        #pragma unroll
        for (int tap = 0; tap < 9; ++tap) {
            const int dy = tap / 3 - 1;
            const int dx = tap % 3 - 1;

            // A fragment (ISA 16-bit A 16x32): lane-half 0 -> K 0-7 & 16-23,
            // lane-half 1 -> K 8-15 & 24-31; each run is 16B contiguous.
            const _Float16* pA = &sW[tap][msub * 16 + lm][hi * 8];
            v8h a0 = *(const v8h*)(pA);
            v8h a1 = *(const v8h*)(pA + 16);
            v16h a = __builtin_shufflevector(a0, a1, 0, 1, 2, 3, 4, 5, 6, 7,
                                             8, 9, 10, 11, 12, 13, 14, 15);

            // B fragments: N = lane%16 (column), K = hi*8 + e and 16 + hi*8 + e (same as A).
            const int col = nsub * 16 + lm + 1 + dx;
            const _Float16* pB0 = &sIn[1 + dy][col][hi * 8];   // row y0
            v8h b00 = *(const v8h*)(pB0);
            v8h b01 = *(const v8h*)(pB0 + 16);
            v16h bf0 = __builtin_shufflevector(b00, b01, 0, 1, 2, 3, 4, 5, 6, 7,
                                               8, 9, 10, 11, 12, 13, 14, 15);
            const _Float16* pB1 = &sIn[2 + dy][col][hi * 8];   // row y0+1
            v8h b10 = *(const v8h*)(pB1);
            v8h b11 = *(const v8h*)(pB1 + 16);
            v16h bf1 = __builtin_shufflevector(b10, b11, 0, 1, 2, 3, 4, 5, 6, 7,
                                               8, 9, 10, 11, 12, 13, 14, 15);

            acc0 = __builtin_amdgcn_wmma_f32_16x16x32_f16(
                false, a, false, bf0, (short)0, acc0, false, false);
            acc1 = __builtin_amdgcn_wmma_f32_16x16x32_f16(
                false, a, false, bf1, (short)0, acc1, false, false);
        }
        __syncthreads();
    }

    // ---- epilogue: C/D layout -> N = lm, M = r + 8*hi ----
    const int gx = x0 + nsub * 16 + lm;
    #pragma unroll
    for (int r = 0; r < 8; ++r) {
        int m = m0 + msub * 16 + r + hi * 8;
        float bv = bias[m];
        float v0 = acc0[r] + bv;
        float v1 = acc1[r] + bv;
        if (do_relu) {
            v0 = v0 > 0.0f ? v0 : 0.0f;
            v1 = v1 > 0.0f ? v1 : 0.0f;
        }
        size_t base = (((size_t)b * Cout + m) * H + y0) * W + gx;
        out[base]     = v0;
        out[base + W] = v1;
    }
}

// ---------------------------------------------------------------------------
// AdaIN: per (b,ch) mean/std (ddof=1) of content+style, then transform.
// One block per channel.
// ---------------------------------------------------------------------------
__global__ __launch_bounds__(256) void wct_kernel(
    const float* __restrict__ c, const float* __restrict__ s,
    float* __restrict__ out, int HW)
{
    size_t base = (size_t)blockIdx.x * HW;
    const float* cp = c + base;
    const float* sp = s + base;

    float c0 = 0.f, c1 = 0.f, s0 = 0.f, s1 = 0.f;
    for (int i = threadIdx.x; i < HW; i += 256) {
        float v = cp[i]; c0 += v; c1 += v * v;
        float u = sp[i]; s0 += u; s1 += u * u;
    }
    __shared__ float red[256];
    __shared__ float stats[4];
    float vals[4] = {c0, c1, s0, s1};
    for (int j = 0; j < 4; ++j) {
        red[threadIdx.x] = vals[j];
        __syncthreads();
        for (int off = 128; off > 0; off >>= 1) {
            if (threadIdx.x < off) red[threadIdx.x] += red[threadIdx.x + off];
            __syncthreads();
        }
        if (threadIdx.x == 0) stats[j] = red[0];
        __syncthreads();
    }
    float n   = (float)HW;
    float cm  = stats[0] / n;
    float cvar = (stats[1] - stats[0] * stats[0] / n) / (n - 1.0f);
    float cs  = sqrtf(cvar > 0.f ? cvar : 0.f);
    float sm  = stats[2] / n;
    float svar = (stats[3] - stats[2] * stats[2] / n) / (n - 1.0f);
    float ss  = sqrtf(svar > 0.f ? svar : 0.f);
    float inv = 1.0f / (cs + 1e-5f);

    float* op = out + base;
    for (int i = threadIdx.x; i < HW; i += 256)
        op[i] = (cp[i] - cm) * inv * ss + sm;
}

// ---------------------------------------------------------------------------
// get_hh: x - nearest_up(avgpool2(x)). One thread per 2x2 quad.
// ---------------------------------------------------------------------------
__global__ __launch_bounds__(256) void get_hh_kernel(
    const float* __restrict__ x, float* __restrict__ out,
    int H, int W, int total_quads)
{
    int i = blockIdx.x * 256 + threadIdx.x;
    if (i >= total_quads) return;
    int qw = W >> 1, qh = H >> 1;
    int qx = i % qw;
    int t  = i / qw;
    int qy = t % qh;
    int bc = t / qh;
    size_t base = (size_t)bc * H * W;
    int y = qy * 2, xx = qx * 2;
    float a = x[base + (size_t)y * W + xx];
    float b = x[base + (size_t)y * W + xx + 1];
    float c = x[base + (size_t)(y + 1) * W + xx];
    float d = x[base + (size_t)(y + 1) * W + xx + 1];
    float m = 0.25f * (a + b + c + d);
    out[base + (size_t)y * W + xx]           = a - m;
    out[base + (size_t)y * W + xx + 1]       = b - m;
    out[base + (size_t)(y + 1) * W + xx]     = c - m;
    out[base + (size_t)(y + 1) * W + xx + 1] = d - m;
}

// ---------------------------------------------------------------------------
// Per-sample masked 9x9 gaussian kernel from alpha. grid = B, block = 128.
// ---------------------------------------------------------------------------
__global__ void gk_kernel(const float* __restrict__ alphas, int col,
                          float* __restrict__ gk)
{
    int b = blockIdx.x;
    int t = threadIdx.x;
    __shared__ float sv[81];
    __shared__ float ssum;
    float val = 0.f;
    if (t < 81) {
        float a = alphas[b * 3 + col];
        float k = floorf(floorf(a * 8.0f) * 0.5f) * 2.0f + 1.0f;
        float r = (k - 1.0f) * 0.5f;
        int cy = t / 9 - 4, cx = t % 9 - 4;
        float d2 = (float)(cy * cy + cx * cx);
        bool inr = (fabsf((float)cy) <= r) && (fabsf((float)cx) <= r);
        val = inr ? expf(-d2 * (1.0f / 512.0f)) : 0.0f;  // 2*sigma^2 = 512
        sv[t] = val;
    }
    __syncthreads();
    if (t == 0) {
        float s = 0.f;
        for (int i = 0; i < 81; ++i) s += sv[i];
        ssum = s;
    }
    __syncthreads();
    if (t < 81) gk[b * 81 + t] = val / ssum;
}

// ---------------------------------------------------------------------------
// out = gaussian_blur(hh) + up2(prev). Replicate pad 4. One thread per pixel.
// ---------------------------------------------------------------------------
__global__ __launch_bounds__(256) void blur_add_kernel(
    const float* __restrict__ hh, const float* __restrict__ gk,
    const float* __restrict__ prev, float* __restrict__ out,
    int C, int H, int W)
{
    size_t idx = (size_t)blockIdx.x * 256 + threadIdx.x;
    int HW = H * W;
    int x = (int)(idx % W);
    size_t t = idx / W;
    int y = (int)(t % H);
    t /= H;
    int c = (int)(t % C);
    int b = (int)(t / C);

    __shared__ float sgk[81];
    int bb = (int)(((size_t)blockIdx.x * 256) / ((size_t)C * HW));
    if (threadIdx.x < 81) sgk[threadIdx.x] = gk[bb * 81 + threadIdx.x];
    __syncthreads();

    const float* p = hh + ((size_t)b * C + c) * HW;
    float acc = 0.f;
    #pragma unroll
    for (int dy = -4; dy <= 4; ++dy) {
        int yy = clampi(y + dy, H);
        #pragma unroll
        for (int dx = -4; dx <= 4; ++dx) {
            int xx = clampi(x + dx, W);
            acc += sgk[(dy + 4) * 9 + (dx + 4)] * p[(size_t)yy * W + xx];
        }
    }
    int hw = W >> 1;
    float pv = prev[((size_t)b * C + c) * ((H >> 1) * hw) +
                    (size_t)(y >> 1) * hw + (x >> 1)];
    out[idx] = acc + pv;
}

// ---------------------------------------------------------------------------
// Final conv: 64 -> 3 channels, reflect pad, no relu. Scalar (Cout too small
// for a WMMA M-tile; ~0.9 GFLOP total).
// ---------------------------------------------------------------------------
__global__ __launch_bounds__(256) void conv_final_kernel(
    const float* __restrict__ in, const float* __restrict__ wgt,
    const float* __restrict__ bias, float* __restrict__ out, int H, int W)
{
    size_t idx = (size_t)blockIdx.x * 256 + threadIdx.x;
    size_t total = (size_t)4 * 3 * H * W;
    if (idx >= total) return;
    int x = (int)(idx % W);
    size_t t = idx / W;
    int y  = (int)(t % H);
    t /= H;
    int co = (int)(t % 3);
    int b  = (int)(t / 3);

    float acc = bias[co];
    for (int ci = 0; ci < 64; ++ci) {
        const float* ip = in + (((size_t)b * 64 + ci) * H) * W;
        const float* wp = wgt + ((size_t)co * 64 + ci) * 9;
        #pragma unroll
        for (int ky = 0; ky < 3; ++ky) {
            int ry = reflect1(y + ky - 1, H);
            #pragma unroll
            for (int kx = 0; kx < 3; ++kx) {
                int rx = reflect1(x + kx - 1, W);
                acc += ip[(size_t)ry * W + rx] * wp[ky * 3 + kx];
            }
        }
    }
    out[idx] = acc;
}

// ---------------------------------------------------------------------------
extern "C" void kernel_launch(void* const* d_in, const int* in_sizes, int n_in,
                              void* d_out, int out_size, void* d_ws, size_t ws_size,
                              hipStream_t stream) {
    const float* x      = (const float*)d_in[0];
    const float* c34    = (const float*)d_in[1];
    const float* c22    = (const float*)d_in[2];
    const float* c12    = (const float*)d_in[3];
    const float* s34    = (const float*)d_in[4];
    const float* s31    = (const float*)d_in[5];
    const float* s22    = (const float*)d_in[6];
    const float* s21    = (const float*)d_in[7];
    const float* s12    = (const float*)d_in[8];
    const float* s11    = (const float*)d_in[9];
    const float* alphas = (const float*)d_in[10];
    const float* w41 = (const float*)d_in[11]; const float* b41 = (const float*)d_in[12];
    const float* w34 = (const float*)d_in[13]; const float* b34 = (const float*)d_in[14];
    const float* w33 = (const float*)d_in[15]; const float* b33 = (const float*)d_in[16];
    const float* w32 = (const float*)d_in[17]; const float* b32 = (const float*)d_in[18];
    const float* w31 = (const float*)d_in[19]; const float* b31 = (const float*)d_in[20];
    const float* w22 = (const float*)d_in[21]; const float* b22 = (const float*)d_in[22];
    const float* w21 = (const float*)d_in[23]; const float* b21 = (const float*)d_in[24];
    const float* w12 = (const float*)d_in[25]; const float* b12 = (const float*)d_in[26];
    const float* w11 = (const float*)d_in[27]; const float* b11 = (const float*)d_in[28];

    const size_t BUF = 16777216ULL;  // 4*64*256*256 floats (largest tensor)
    float* A  = (float*)d_ws;
    float* Bb = A + BUF;
    float* Cc = Bb + BUF;
    float* GK = Cc + BUF;                       // 4*81 floats
    _Float16* P = (_Float16*)(GK + 1024);       // packed f16 weights (<=1.18M halfs)
    float* out = (float*)d_out;

    // ---- level 4 -> 3 ----
    // t41 = relu(conv(x, w41))  [4,256,32,32] -> A
    pack_w_kernel<<<dim3((256 * 512 * 9 + 255) / 256), 256, 0, stream>>>(w41, P, 256, 512);
    conv3x3_wmma_kernel<<<dim3(16, 4, 4), 256, 0, stream>>>(x, P, b41, A, 512, 256, 32, 32, 1);
    // wct34 = wct(c34, s34) -> Bb
    wct_kernel<<<dim3(4 * 256), 256, 0, stream>>>(c34, s34, Bb, 64 * 64);
    // hh34 -> Cc
    get_hh_kernel<<<dim3((4 * 256 * 32 * 32 + 255) / 256), 256, 0, stream>>>(Bb, Cc, 64, 64, 4 * 256 * 32 * 32);
    gk_kernel<<<dim3(4), 128, 0, stream>>>(alphas, 2, GK);
    // a34 = blur(hh34) + up2(t41) -> Bb
    blur_add_kernel<<<dim3(4 * 256 * 64 * 64 / 256), 256, 0, stream>>>(Cc, GK, A, Bb, 256, 64, 64);
    // conv chain at 64x64
    pack_w_kernel<<<dim3((256 * 256 * 9 + 255) / 256), 256, 0, stream>>>(w34, P, 256, 256);
    conv3x3_wmma_kernel<<<dim3(2 * 32, 4, 4), 256, 0, stream>>>(Bb, P, b34, A, 256, 256, 64, 64, 1);
    pack_w_kernel<<<dim3((256 * 256 * 9 + 255) / 256), 256, 0, stream>>>(w33, P, 256, 256);
    conv3x3_wmma_kernel<<<dim3(2 * 32, 4, 4), 256, 0, stream>>>(A, P, b33, Cc, 256, 256, 64, 64, 1);
    pack_w_kernel<<<dim3((256 * 256 * 9 + 255) / 256), 256, 0, stream>>>(w32, P, 256, 256);
    conv3x3_wmma_kernel<<<dim3(2 * 32, 4, 4), 256, 0, stream>>>(Cc, P, b32, A, 256, 256, 64, 64, 1);
    wct_kernel<<<dim3(4 * 256), 256, 0, stream>>>(A, s31, Bb, 64 * 64);
    // t31 = relu(conv(.., w31))  [4,128,64,64] -> A
    pack_w_kernel<<<dim3((128 * 256 * 9 + 255) / 256), 256, 0, stream>>>(w31, P, 128, 256);
    conv3x3_wmma_kernel<<<dim3(2 * 32, 2, 4), 256, 0, stream>>>(Bb, P, b31, A, 256, 128, 64, 64, 1);

    // ---- level 3 -> 2 ----
    wct_kernel<<<dim3(4 * 128), 256, 0, stream>>>(c22, s22, Cc, 128 * 128);
    get_hh_kernel<<<dim3((4 * 128 * 64 * 64 + 255) / 256), 256, 0, stream>>>(Cc, Bb, 128, 128, 4 * 128 * 64 * 64);
    gk_kernel<<<dim3(4), 128, 0, stream>>>(alphas, 1, GK);
    blur_add_kernel<<<dim3(4 * 128 * 128 * 128 / 256), 256, 0, stream>>>(Bb, GK, A, Cc, 128, 128, 128);
    pack_w_kernel<<<dim3((128 * 128 * 9 + 255) / 256), 256, 0, stream>>>(w22, P, 128, 128);
    conv3x3_wmma_kernel<<<dim3(4 * 64, 2, 4), 256, 0, stream>>>(Cc, P, b22, Bb, 128, 128, 128, 128, 1);
    wct_kernel<<<dim3(4 * 128), 256, 0, stream>>>(Bb, s21, Cc, 128 * 128);
    // t21 = relu(conv(.., w21))  [4,64,128,128] -> A
    pack_w_kernel<<<dim3((64 * 128 * 9 + 255) / 256), 256, 0, stream>>>(w21, P, 64, 128);
    conv3x3_wmma_kernel<<<dim3(4 * 64, 1, 4), 256, 0, stream>>>(Cc, P, b21, A, 128, 64, 128, 128, 1);

    // ---- level 2 -> 1 ----
    wct_kernel<<<dim3(4 * 64), 256, 0, stream>>>(c12, s12, Bb, 256 * 256);
    get_hh_kernel<<<dim3((4 * 64 * 128 * 128 + 255) / 256), 256, 0, stream>>>(Bb, Cc, 256, 256, 4 * 64 * 128 * 128);
    gk_kernel<<<dim3(4), 128, 0, stream>>>(alphas, 0, GK);
    blur_add_kernel<<<dim3(4 * 64 * 256 * 256 / 256), 256, 0, stream>>>(Cc, GK, A, Bb, 64, 256, 256);
    pack_w_kernel<<<dim3((64 * 64 * 9 + 255) / 256), 256, 0, stream>>>(w12, P, 64, 64);
    conv3x3_wmma_kernel<<<dim3(8 * 128, 1, 4), 256, 0, stream>>>(Bb, P, b12, A, 64, 64, 256, 256, 1);
    wct_kernel<<<dim3(4 * 64), 256, 0, stream>>>(A, s11, Bb, 256 * 256);
    // final conv -> d_out  [4,3,256,256]
    conv_final_kernel<<<dim3((4 * 3 * 256 * 256 + 255) / 256), 256, 0, stream>>>(Bb, w11, b11, out, 256, 256);
}
